// MoE_14087492731074
// MI455X (gfx1250) — compile-verified, matrix operand order checked by reading, not run on hardware
//
#include <hip/hip_runtime.h>
#include <stdint.h>

#define N_TOK 8192
#define DIM   1024
#define HID   4096
#define OUTD  1024
#define NEXP  8
#define TOPK  2
#define CAPE  4096

typedef __bf16 bf16;
typedef __attribute__((ext_vector_type(16))) __bf16 v16bf;
typedef __attribute__((ext_vector_type(8)))  float  v8f;

__device__ __forceinline__ uint16_t f2bf_bits(float f) {
    union { float f; uint32_t u; } a; a.f = f;
    uint32_t u = a.u;
    uint32_t r = u + 0x7FFFu + ((u >> 16) & 1u);   // round-to-nearest-even
    return (uint16_t)(r >> 16);
}

// async DMA: global -> LDS, 16 bytes per lane, tracked by ASYNCcnt
__device__ __forceinline__ void async_ld_b128(uint32_t lds_off, uint32_t voff,
                                              uint64_t sbase) {
    asm volatile("global_load_async_to_lds_b128 %0, %1, %2 offset:0"
                 :: "v"(lds_off), "v"(voff), "s"(sbase) : "memory");
}
__device__ __forceinline__ void wait_asynccnt0() {
    asm volatile("s_wait_asynccnt 0x0" ::: "memory");
}

// ---------------------------------------------------------------------------
// 0) zero dispatch tables (empty slots must point at token 0 with weight 0)
// ---------------------------------------------------------------------------
__global__ void moe_zero(int* __restrict__ tokA, float* __restrict__ dispw) {
    int i = blockIdx.x * blockDim.x + threadIdx.x;   // grid covers E*CAPE
    tokA[i]  = 0;
    dispw[i] = 0.f;
}

// ---------------------------------------------------------------------------
// 1) gating: logits, softmax, top-2
// ---------------------------------------------------------------------------
__global__ __launch_bounds__(256)
void moe_gate(const float* __restrict__ x, const float* __restrict__ gateW,
              const float* __restrict__ gateB,
              float* __restrict__ probs, int* __restrict__ tki,
              float* __restrict__ tkp) {
    __shared__ float gw[NEXP * DIM];                 // 32 KB
    const int tid = threadIdx.x;
    for (int i = tid; i < NEXP * DIM; i += 256) gw[i] = gateW[i];
    __syncthreads();

    const int n = blockIdx.x * 256 + tid;
    float acc[NEXP];
    #pragma unroll
    for (int e = 0; e < NEXP; ++e) acc[e] = gateB[e];

    const float4* xr = (const float4*)(x + (size_t)n * DIM);
    for (int d4 = 0; d4 < DIM / 4; ++d4) {
        float4 xv = xr[d4];
        #pragma unroll
        for (int e = 0; e < NEXP; ++e) {
            float4 wv = *(const float4*)(&gw[e * DIM + d4 * 4]);
            acc[e] += xv.x * wv.x + xv.y * wv.y + xv.z * wv.z + xv.w * wv.w;
        }
    }
    float mx = acc[0];
    #pragma unroll
    for (int e = 1; e < NEXP; ++e) mx = fmaxf(mx, acc[e]);
    float p[NEXP], s = 0.f;
    #pragma unroll
    for (int e = 0; e < NEXP; ++e) { p[e] = expf(acc[e] - mx); s += p[e]; }
    float inv = 1.f / s;
    #pragma unroll
    for (int e = 0; e < NEXP; ++e) { p[e] *= inv; probs[(size_t)n * NEXP + e] = p[e]; }

    int i1 = 0; float v1 = p[0];
    #pragma unroll
    for (int e = 1; e < NEXP; ++e) if (p[e] > v1) { v1 = p[e]; i1 = e; }
    int i2 = -1; float v2 = -1.f;
    #pragma unroll
    for (int e = 0; e < NEXP; ++e) if (e != i1 && p[e] > v2) { v2 = p[e]; i2 = e; }

    tki[n * 2 + 0] = i1;  tki[n * 2 + 1] = i2;
    tkp[n * 2 + 0] = v1;  tkp[n * 2 + 1] = v2;
}

// ---------------------------------------------------------------------------
// 2) aux loss: deterministic single-block reduction
// ---------------------------------------------------------------------------
__device__ float block_reduce_sum(float v, float* sbuf) {
    const int lane = threadIdx.x & 31, wid = threadIdx.x >> 5;
    #pragma unroll
    for (int o = 16; o > 0; o >>= 1) v += __shfl_down(v, o, 32);
    if (lane == 0) sbuf[wid] = v;
    __syncthreads();
    float r = 0.f;
    if (wid == 0) {
        r = sbuf[lane];
        #pragma unroll
        for (int o = 16; o > 0; o >>= 1) r += __shfl_down(r, o, 32);
    }
    __syncthreads();
    return r;                                        // valid on thread 0
}

__global__ __launch_bounds__(1024)
void moe_stats(const float* __restrict__ probs, const int* __restrict__ tki,
               float* __restrict__ out) {
    __shared__ float sbuf[32];
    const int tid = threadIdx.x;
    float imp[NEXP] = {}, rout[NEXP] = {}, cnt[NEXP] = {};
    for (int n = tid; n < N_TOK; n += 1024) {
        float p[NEXP];
        #pragma unroll
        for (int e = 0; e < NEXP; ++e) { p[e] = probs[(size_t)n * NEXP + e]; imp[e] += p[e]; }
        int i1 = tki[n * 2], i2 = tki[n * 2 + 1];
        cnt[i1] += 1.f; rout[i1] += p[i1];
        cnt[i2] += 1.f; rout[i2] += p[i2];
    }
    float impS[NEXP], cntS[NEXP], routS[NEXP];
    for (int e = 0; e < NEXP; ++e) impS[e]  = block_reduce_sum(imp[e],  sbuf);
    for (int e = 0; e < NEXP; ++e) cntS[e]  = block_reduce_sum(cnt[e],  sbuf);
    for (int e = 0; e < NEXP; ++e) routS[e] = block_reduce_sum(rout[e], sbuf);
    if (tid == 0) {
        float mean = 0.f;
        for (int e = 0; e < NEXP; ++e) mean += impS[e];
        mean /= NEXP;
        float var = 0.f;
        for (int e = 0; e < NEXP; ++e) { float d = impS[e] - mean; var += d * d; }
        var /= (NEXP - 1);
        float il = var / (float)(NEXP * NEXP);
        float lb = 0.f;
        for (int e = 0; e < NEXP; ++e)
            lb += (cntS[e] / (float)N_TOK) * (routS[e] / (float)N_TOK);
        lb *= (float)NEXP;
        out[(size_t)N_TOK * OUTD] = il + lb;
    }
}

// ---------------------------------------------------------------------------
// 3) dispatch: flat-order stable prefix count per expert (capacity CAPE)
// ---------------------------------------------------------------------------
__global__ __launch_bounds__(1024)
void moe_dispatch(const int* __restrict__ tki, const float* __restrict__ tkp,
                  int* __restrict__ tokA, float* __restrict__ dispw,
                  int* __restrict__ slotOf) {
    __shared__ int buf[1024 * NEXP];                 // 32 KB
    const int t = threadIdx.x;
    const int ITEMS = (N_TOK * TOPK) / 1024;         // 16 per thread, flat order
    int eloc[16]; float ploc[16];
    int cnt[NEXP] = {};
    #pragma unroll
    for (int it = 0; it < ITEMS; ++it) {
        int item = t * ITEMS + it;
        int ee = tki[item];
        eloc[it] = ee; ploc[it] = tkp[item];
        cnt[ee]++;
    }
    int cur[NEXP];
    #pragma unroll
    for (int e = 0; e < NEXP; ++e) { cur[e] = cnt[e]; buf[t * NEXP + e] = cnt[e]; }
    __syncthreads();
    for (int off = 1; off < 1024; off <<= 1) {       // Hillis-Steele 8-vec scan
        int oth[NEXP] = {};
        if (t >= off) {
            #pragma unroll
            for (int e = 0; e < NEXP; ++e) oth[e] = buf[(t - off) * NEXP + e];
        }
        __syncthreads();
        #pragma unroll
        for (int e = 0; e < NEXP; ++e) { cur[e] += oth[e]; buf[t * NEXP + e] = cur[e]; }
        __syncthreads();
    }
    int run[NEXP];
    #pragma unroll
    for (int e = 0; e < NEXP; ++e) run[e] = cur[e] - cnt[e];   // exclusive prefix
    #pragma unroll
    for (int it = 0; it < ITEMS; ++it) {
        int item = t * ITEMS + it;
        int ee = eloc[it];
        int pos = run[ee]++;
        if (pos < CAPE) {
            int slot = ee * CAPE + pos;
            tokA[slot]  = item >> 1;                 // token id
            dispw[slot] = ploc[it];
            slotOf[item] = slot;
        } else {
            slotOf[item] = -1;                       // dropped
        }
    }
}

// ---------------------------------------------------------------------------
// 3b) weight pretranspose: W (E,K,N) f32 -> Wt (E,N,K) bf16 (N-major for WMMA B)
// ---------------------------------------------------------------------------
__global__ __launch_bounds__(1024)
void moe_transpose(const float* __restrict__ W, bf16* __restrict__ Wt,
                   int Kfull, int Nfull) {
    __shared__ float t[32][33];
    const int e  = blockIdx.z;
    const int kk = blockIdx.y * 32 + threadIdx.y;
    const int nn = blockIdx.x * 32 + threadIdx.x;
    t[threadIdx.y][threadIdx.x] = W[((size_t)e * Kfull + kk) * Nfull + nn];
    __syncthreads();
    const int ko = blockIdx.y * 32 + threadIdx.x;
    const int no = blockIdx.x * 32 + threadIdx.y;
    ((uint16_t*)Wt)[((size_t)e * Nfull + no) * Kfull + ko] =
        f2bf_bits(t[threadIdx.x][threadIdx.y]);
}

// ---------------------------------------------------------------------------
// 4) layer 1 GEMM: h[e] = relu(x[tok] @ W1[e] + b1[e]), bf16 WMMA
//    block tile 128(M) x 256(N) x 32(K); 8 waves, each wave 64x64 (4x4 tiles)
//    A: gathered f32 rows, VGPR convert -> LDS;  B: async DMA global->LDS
// ---------------------------------------------------------------------------
#define LDT 40                                       // bf16 row stride (80 B, 16B-aligned)

union FragBF { v16bf v; uint4 q[2]; };

__global__ __launch_bounds__(256)
void moe_gemm1(const float* __restrict__ x, const bf16* __restrict__ W1t,
               const float* __restrict__ b1, const int* __restrict__ tokA,
               bf16* __restrict__ h) {
    const int e  = blockIdx.z;
    const int n0 = blockIdx.x * 256;
    const int m0 = blockIdx.y * 128;
    const int tid  = threadIdx.x;
    const int lane = tid & 31;
    const int wid  = tid >> 5;
    const int wm = wid & 1, wn = wid >> 1;           // 2 waves M x 4 waves N
    const int lr = lane & 15, lh = lane >> 4;

    __shared__ __align__(16) bf16 As[128 * LDT];     // M x K, row-major
    __shared__ __align__(16) bf16 Bs[256 * LDT];     // N x K (N-major)
    __shared__ int tokS[128];

    if (tid < 128) tokS[tid] = tokA[e * CAPE + m0 + tid];
    __syncthreads();

    v8f acc[4][4];
    #pragma unroll
    for (int i = 0; i < 4; ++i)
        #pragma unroll
        for (int j = 0; j < 4; ++j) acc[i][j] = (v8f)0.f;

    // per-thread loop-invariant staging addresses
    int ar[4], ag[4];
    uint32_t bofs[4], blds[4];
    #pragma unroll
    for (int i = 0; i < 4; ++i) {
        int idx = tid + 256 * i;
        ar[i] = idx >> 3; ag[i] = idx & 7;           // A: 128 rows x 8 float4 groups
        int bn = idx >> 2, bg = idx & 3;             // B: 256 N rows x 4 groups of 8 bf16
        bofs[i] = (uint32_t)((bn * DIM + 8 * bg) * 2);
        blds[i] = (uint32_t)(uintptr_t)&Bs[bn * LDT + 8 * bg];
    }
    const uint16_t* wp = (const uint16_t*)W1t;

    for (int k0 = 0; k0 < DIM; k0 += 32) {
        // A: gather x rows into regs (overlaps with previous compute)
        float4 av[4];
        #pragma unroll
        for (int i = 0; i < 4; ++i)
            av[i] = *(const float4*)(x + (size_t)tokS[ar[i]] * DIM + k0 + 4 * ag[i]);

        __syncthreads();                             // prior iter done reading LDS

        // B: async DMA straight into LDS (no VGPR round-trip)
        uint64_t bbase = (uint64_t)(uintptr_t)(wp + (size_t)(e * HID + n0) * DIM + k0);
        #pragma unroll
        for (int i = 0; i < 4; ++i) async_ld_b128(blds[i], bofs[i], bbase);

        // A: f32 -> bf16, packed b64 store
        #pragma unroll
        for (int i = 0; i < 4; ++i) {
            union { uint16_t s[4]; uint2 u; } pk;
            pk.s[0] = f2bf_bits(av[i].x); pk.s[1] = f2bf_bits(av[i].y);
            pk.s[2] = f2bf_bits(av[i].z); pk.s[3] = f2bf_bits(av[i].w);
            *(uint2*)&As[ar[i] * LDT + 4 * ag[i]] = pk.u;
        }
        wait_asynccnt0();
        __syncthreads();

        FragBF af[4], bf_[4];
        #pragma unroll
        for (int i = 0; i < 4; ++i) {
            const bf16* p = &As[(64 * wm + 16 * i + lr) * LDT + 8 * lh];
            af[i].q[0] = *(const uint4*)p;           // K 0..7 (or 8..15)
            af[i].q[1] = *(const uint4*)(p + 16);    // K 16..23 (or 24..31)
        }
        #pragma unroll
        for (int j = 0; j < 4; ++j) {
            const bf16* p = &Bs[(64 * wn + 16 * j + lr) * LDT + 8 * lh];
            bf_[j].q[0] = *(const uint4*)p;
            bf_[j].q[1] = *(const uint4*)(p + 16);
        }
        #pragma unroll
        for (int i = 0; i < 4; ++i)
            #pragma unroll
            for (int j = 0; j < 4; ++j)
                acc[i][j] = __builtin_amdgcn_wmma_f32_16x16x32_bf16(
                    false, af[i].v, false, bf_[j].v, (short)0, acc[i][j], false, false);
    }

    // epilogue: bias + relu -> bf16 h
    uint16_t* hp = (uint16_t*)h;
    #pragma unroll
    for (int i = 0; i < 4; ++i) {
        #pragma unroll
        for (int j = 0; j < 4; ++j) {
            int gn = n0 + 64 * wn + 16 * j + lr;
            float bias = b1[e * HID + gn];
            #pragma unroll
            for (int v = 0; v < 8; ++v) {
                int gm = m0 + 64 * wm + 16 * i + 8 * lh + v;
                float val = acc[i][j][v] + bias;
                val = val > 0.f ? val : 0.f;
                hp[(size_t)(e * CAPE + gm) * HID + gn] = f2bf_bits(val);
            }
        }
    }
}

// ---------------------------------------------------------------------------
// 5) layer 2 GEMM: yws[e] = (h[e] @ W2[e] + b2[e]) * dispw, bf16 WMMA
//    A and B tiles both bf16: all staging via async DMA global->LDS
// ---------------------------------------------------------------------------
__global__ __launch_bounds__(256)
void moe_gemm2(const bf16* __restrict__ h, const bf16* __restrict__ W2t,
               const float* __restrict__ b2, const float* __restrict__ dispw,
               float* __restrict__ yws) {
    const int e  = blockIdx.z;
    const int n0 = blockIdx.x * 256;
    const int m0 = blockIdx.y * 128;
    const int tid  = threadIdx.x;
    const int lane = tid & 31;
    const int wid  = tid >> 5;
    const int wm = wid & 1, wn = wid >> 1;
    const int lr = lane & 15, lh = lane >> 4;

    __shared__ __align__(16) bf16 As[128 * LDT];
    __shared__ __align__(16) bf16 Bs[256 * LDT];
    __shared__ float wS[128];

    if (tid < 128) wS[tid] = dispw[e * CAPE + m0 + tid];
    __syncthreads();

    v8f acc[4][4];
    #pragma unroll
    for (int i = 0; i < 4; ++i)
        #pragma unroll
        for (int j = 0; j < 4; ++j) acc[i][j] = (v8f)0.f;

    // loop-invariant staging addresses
    uint32_t aofs[2], alds[2], bofs[4], blds[4];
    #pragma unroll
    for (int i = 0; i < 2; ++i) {
        int idx = tid + 256 * i;                     // A: 128 rows x 4 groups of 8 bf16
        int r = idx >> 2, g = idx & 3;
        aofs[i] = (uint32_t)((r * HID + 8 * g) * 2);
        alds[i] = (uint32_t)(uintptr_t)&As[r * LDT + 8 * g];
    }
    #pragma unroll
    for (int i = 0; i < 4; ++i) {
        int idx = tid + 256 * i;                     // B: 256 N rows x 4 groups
        int bn = idx >> 2, bg = idx & 3;
        bofs[i] = (uint32_t)((bn * HID + 8 * bg) * 2);
        blds[i] = (uint32_t)(uintptr_t)&Bs[bn * LDT + 8 * bg];
    }
    const uint16_t* hp = (const uint16_t*)h;
    const uint16_t* wp = (const uint16_t*)W2t;

    for (int k0 = 0; k0 < HID; k0 += 32) {
        __syncthreads();                             // prior iter done reading LDS
        uint64_t abase = (uint64_t)(uintptr_t)(hp + (size_t)(e * CAPE + m0) * HID + k0);
        uint64_t bbase = (uint64_t)(uintptr_t)(wp + (size_t)(e * OUTD + n0) * HID + k0);
        #pragma unroll
        for (int i = 0; i < 2; ++i) async_ld_b128(alds[i], aofs[i], abase);
        #pragma unroll
        for (int i = 0; i < 4; ++i) async_ld_b128(blds[i], bofs[i], bbase);
        wait_asynccnt0();
        __syncthreads();

        FragBF af[4], bf_[4];
        #pragma unroll
        for (int i = 0; i < 4; ++i) {
            const bf16* p = &As[(64 * wm + 16 * i + lr) * LDT + 8 * lh];
            af[i].q[0] = *(const uint4*)p;
            af[i].q[1] = *(const uint4*)(p + 16);
        }
        #pragma unroll
        for (int j = 0; j < 4; ++j) {
            const bf16* p = &Bs[(64 * wn + 16 * j + lr) * LDT + 8 * lh];
            bf_[j].q[0] = *(const uint4*)p;
            bf_[j].q[1] = *(const uint4*)(p + 16);
        }
        #pragma unroll
        for (int i = 0; i < 4; ++i)
            #pragma unroll
            for (int j = 0; j < 4; ++j)
                acc[i][j] = __builtin_amdgcn_wmma_f32_16x16x32_bf16(
                    false, af[i].v, false, bf_[j].v, (short)0, acc[i][j], false, false);
    }

    #pragma unroll
    for (int i = 0; i < 4; ++i) {
        #pragma unroll
        for (int j = 0; j < 4; ++j) {
            int gn = n0 + 64 * wn + 16 * j + lr;
            float bias = b2[e * OUTD + gn];
            #pragma unroll
            for (int v = 0; v < 8; ++v) {
                int lm = 64 * wm + 16 * i + 8 * lh + v;
                int gm = m0 + lm;
                yws[(size_t)(e * CAPE + gm) * OUTD + gn] = (acc[i][j][v] + bias) * wS[lm];
            }
        }
    }
}

// ---------------------------------------------------------------------------
// 6) combine: out[n] = sum over this token's kept slots (atomic-free gather)
// ---------------------------------------------------------------------------
__global__ __launch_bounds__(256)
void moe_gather(const float* __restrict__ yws, const int* __restrict__ slotOf,
                float* __restrict__ out) {
    int idx = blockIdx.x * 256 + threadIdx.x;        // over N*OUTD/4
    int n  = idx / (OUTD / 4);
    int o4 = idx % (OUTD / 4);
    float4 r; r.x = r.y = r.z = r.w = 0.f;
    int s0 = slotOf[n * 2 + 0];
    int s1 = slotOf[n * 2 + 1];
    if (s0 >= 0) {
        float4 t = *(const float4*)(yws + (size_t)s0 * OUTD + 4 * o4);
        r.x += t.x; r.y += t.y; r.z += t.z; r.w += t.w;
    }
    if (s1 >= 0) {
        float4 t = *(const float4*)(yws + (size_t)s1 * OUTD + 4 * o4);
        r.x += t.x; r.y += t.y; r.z += t.z; r.w += t.w;
    }
    *(float4*)(out + (size_t)n * OUTD + 4 * o4) = r;
}

// ---------------------------------------------------------------------------
extern "C" void kernel_launch(void* const* d_in, const int* in_sizes, int n_in,
                              void* d_out, int out_size, void* d_ws, size_t ws_size,
                              hipStream_t stream) {
    const float* x      = (const float*)d_in[0];
    const float* gate_W = (const float*)d_in[1];
    const float* gate_b = (const float*)d_in[2];
    const float* W1     = (const float*)d_in[3];
    const float* b1     = (const float*)d_in[4];
    const float* W2     = (const float*)d_in[5];
    const float* b2     = (const float*)d_in[6];
    float* out = (float*)d_out;

    char* ws = (char*)d_ws;
    float* probs  = (float*)(ws + 0);                             // 256 KB
    int*   tki    = (int*)  (ws + (size_t)262144);                // 64 KB
    float* tkp    = (float*)(ws + (size_t)327680);                // 64 KB
    int*   tokA   = (int*)  (ws + (size_t)393216);                // 128 KB
    float* dispw  = (float*)(ws + (size_t)524288);                // 128 KB
    int*   slotOf = (int*)  (ws + (size_t)655360);                // 64 KB
    // W2t: 64 MB bf16, live until gemm2
    bf16*  W2t    = (bf16*) (ws + (size_t)1048576);
    // h: 256 MB bf16, live gemm1 -> gemm2
    bf16*  h      = (bf16*) (ws + (size_t)1048576 + (size_t)67108864);
    // region R: W1t (64 MB bf16, dead after gemm1) aliased with yws (128 MB f32)
    char*  R      = ws + (size_t)1048576 + (size_t)67108864 + (size_t)268435456;
    bf16*  W1t    = (bf16*)R;
    float* yws    = (float*)R;

    moe_zero<<<(NEXP * CAPE) / 256, 256, 0, stream>>>(tokA, dispw);
    moe_gate<<<N_TOK / 256, 256, 0, stream>>>(x, gate_W, gate_b, probs, tki, tkp);
    moe_stats<<<1, 1024, 0, stream>>>(probs, tki, out);
    moe_dispatch<<<1, 1024, 0, stream>>>(tki, tkp, tokA, dispw, slotOf);

    dim3 tb(32, 32);
    dim3 tg1(HID / 32, DIM / 32, NEXP);                           // W1 (E,D,H)
    moe_transpose<<<tg1, tb, 0, stream>>>(W1, W1t, DIM, HID);
    dim3 tg2(OUTD / 32, HID / 32, NEXP);                          // W2 (E,H,O)
    moe_transpose<<<tg2, tb, 0, stream>>>(W2, W2t, HID, OUTD);

    dim3 g1(HID / 256, CAPE / 128, NEXP);                         // 16x32x8 blocks
    moe_gemm1<<<g1, 256, 0, stream>>>(x, W1t, b1, tokA, h);
    dim3 g2(OUTD / 256, CAPE / 128, NEXP);                        // 4x32x8 blocks
    moe_gemm2<<<g2, 256, 0, stream>>>(h, W2t, b2, dispw, yws);

    moe_gather<<<(N_TOK * OUTD / 4) / 256, 256, 0, stream>>>(yws, slotOf, out);
}
